// MoeLayer_35802847379943
// MI455X (gfx1250) — compile-verified, matrix operand order checked by reading, not run on hardware
//
#include <hip/hip_runtime.h>

#define T_TOK 8192
#define D_DIM 1024
#define F_DIM 3584
#define E_EXP 8

#define BM 32          // tokens per workgroup tile (2 WMMA row-tiles)
#define FC 128         // F-chunk per outer iteration (8 waves * 16)
#define XS_STRIDE (D_DIM + 16)   // bf16 elems, rows 32B-aligned
#define HS_STRIDE (FC + 16)
#define W2_STRIDE 32             // W2sT row: 32 k-values (64B)

typedef __attribute__((ext_vector_type(16))) __bf16 v16bf;
typedef __attribute__((ext_vector_type(2)))  __bf16 v2bf;
typedef __attribute__((ext_vector_type(8)))  float  v8f;
typedef __attribute__((ext_vector_type(4)))  unsigned int u32x4;
typedef __attribute__((ext_vector_type(8)))  int i32x8;
typedef __attribute__((ext_vector_type(4)))  int i32x4;

// dynamic LDS footprint of moe_ffn_kernel (bytes)
#define SMEM_BYTES ((size_t)(BM * XS_STRIDE + BM * HS_STRIDE + D_DIM * W2_STRIDE) * 2 \
                    + BM * 4 + BM * 4)

// Load 16 contiguous fp32 and convert to a bf16 fragment half-vector.
__device__ inline v16bf load16_row_bf16(const float* __restrict__ p) {
    v16bf r;
    #pragma unroll
    for (int i = 0; i < 16; i += 4) {
        const float4 f = *(const float4*)(p + i);
        r[i + 0] = (__bf16)f.x;
        r[i + 1] = (__bf16)f.y;
        r[i + 2] = (__bf16)f.z;
        r[i + 3] = (__bf16)f.w;
    }
    return r;
}

// ---------------- gating: logits -> top-2 -> softmax -> per-expert lists ----
__global__ __launch_bounds__(256) void moe_gate_kernel(
    const float* __restrict__ x,
    const float* __restrict__ gw,
    int*   __restrict__ cnt,
    int*   __restrict__ idxl,
    float* __restrict__ coefl)
{
    __shared__ float gws[E_EXP * D_DIM];   // 32 KB
    const int tid = threadIdx.x;

    // TDM: wave 0 DMA-loads gate_w (byte-exact fp32, 32KB) straight into LDS.
    // Descriptor per ISA 8.3/8.4: count=1, data_size=4B, tile = 1 row x 8192 elems.
    if (__builtin_amdgcn_readfirstlane(tid) == 0) {   // scalar-uniform branch (TDM ignores EXEC)
        const unsigned long long ga = (unsigned long long)(uintptr_t)gw;
        const unsigned int lds = (unsigned int)(uintptr_t)(void*)gws; // LDS byte offset
        const unsigned int n = E_EXP * D_DIM;                          // 8192 elements
        u32x4 g0;
        g0[0] = 1u;                                            // count=1 (user descriptor)
        g0[1] = lds;                                           // lds_addr
        g0[2] = (unsigned int)(ga & 0xFFFFFFFFu);              // global_addr[31:0]
        g0[3] = (unsigned int)((ga >> 32) & 0x1FFFFFFu) | (2u << 30); // addr[56:32]|type=2
        i32x8 g1;
        g1[0] = (int)(2u << 16);                 // wg_mask=0, data_size=2 (4 bytes)
        g1[1] = (int)((n & 0xFFFFu) << 16);      // tensor_dim0[15:0]
        g1[2] = (int)((n >> 16) | (1u << 16));   // tensor_dim0[31:16] | tensor_dim1=1
        g1[3] = (int)((n & 0xFFFFu) << 16);      // tile_dim0 = n
        g1[4] = 1;                               // tile_dim1=1, tile_dim2=0
        g1[5] = (int)n;                          // tensor_dim0_stride[31:0]
        g1[6] = 0;
        g1[7] = 0;
        const i32x4 gz4 = (i32x4){};
        const i32x8 gz8 = (i32x8){};
        __builtin_amdgcn_tensor_load_to_lds(g0, g1, gz4, gz4, gz8, 0);
        __builtin_amdgcn_s_wait_tensorcnt(0);
    }
    __syncthreads();

    const int t = blockIdx.x * 256 + tid;
    float acc[E_EXP];
    #pragma unroll
    for (int e = 0; e < E_EXP; ++e) acc[e] = 0.f;

    const float* xr = x + (size_t)t * D_DIM;
    for (int d = 0; d < D_DIM; ++d) {
        const float xv = xr[d];
        #pragma unroll
        for (int e = 0; e < E_EXP; ++e)
            acc[e] = fmaf(xv, gws[e * D_DIM + d], acc[e]);
    }

    int   i0 = 0; float v0 = acc[0];
    #pragma unroll
    for (int e = 1; e < E_EXP; ++e) if (acc[e] > v0) { v0 = acc[e]; i0 = e; }
    int   i1 = -1; float v1 = -3.4e38f;
    #pragma unroll
    for (int e = 0; e < E_EXP; ++e) if (e != i0 && acc[e] > v1) { v1 = acc[e]; i1 = e; }

    const float w0 = 1.f / (1.f + __expf(v1 - v0));   // softmax over {v0, v1}
    const float w1 = 1.f - w0;

    int p0 = atomicAdd(&cnt[i0], 1);
    idxl [i0 * T_TOK + p0] = t;
    coefl[i0 * T_TOK + p0] = w0;
    int p1 = atomicAdd(&cnt[i1], 1);
    idxl [i1 * T_TOK + p1] = t;
    coefl[i1 * T_TOK + p1] = w1;
}

// ---------------- fused routed SwiGLU FFN with bf16 WMMA --------------------
__global__ __launch_bounds__(256) void moe_ffn_kernel(
    const float* __restrict__ x,
    const float* __restrict__ w1,
    const float* __restrict__ w2,
    const float* __restrict__ w3,
    const int*   __restrict__ cnt,
    const int*   __restrict__ idxl,
    const float* __restrict__ coefl,
    float* __restrict__ out)
{
    const int e    = blockIdx.y;
    const int tile = blockIdx.x;
    const int nrow = cnt[e];
    if (tile * BM >= nrow) return;

    extern __shared__ __align__(32) char smem_raw[];
    __bf16* Xs   = (__bf16*)smem_raw;               // [BM][XS_STRIDE]
    __bf16* Hs   = Xs + BM * XS_STRIDE;             // [BM][HS_STRIDE]
    __bf16* W2sT = Hs + BM * HS_STRIDE;             // [D_DIM][W2_STRIDE] transposed slab
    int*    tok_s  = (int*)(W2sT + D_DIM * W2_STRIDE);
    float*  coef_s = (float*)(tok_s + BM);

    const int tid  = threadIdx.x;
    const int lane = tid & 31;
    const int wv   = tid >> 5;        // wave id 0..7, owns out cols [wv*128, wv*128+128)
    const int half = lane >> 4;       // k-half select (A/B frags)
    const int lid  = lane & 15;       // row (A) / col (B) / n (C,D)

    if (tid < BM) {
        const int g = tile * BM + tid;
        if (g < nrow) {
            tok_s[tid]  = idxl [e * T_TOK + g];
            coef_s[tid] = coefl[e * T_TOK + g];
        } else {                      // pad rows: valid address, zero coefficient
            tok_s[tid]  = idxl[e * T_TOK + tile * BM];
            coef_s[tid] = 0.f;
        }
    }
    __syncthreads();

    // stage X tile (fp32 -> bf16) into LDS, coalesced
    for (int i = tid; i < BM * D_DIM; i += 256) {
        const int r = i >> 10;
        const int c = i & (D_DIM - 1);
        Xs[r * XS_STRIDE + c] = (__bf16)x[(size_t)tok_s[r] * D_DIM + c];
    }
    __syncthreads();

    v8f accO[2][8];
    #pragma unroll
    for (int r2 = 0; r2 < 2; ++r2)
        #pragma unroll
        for (int j = 0; j < 8; ++j) accO[r2][j] = (v8f){};

    const size_t ebase = (size_t)e * F_DIM * D_DIM;

    for (int fc = 0; fc < F_DIM; fc += FC) {
        // ---- stage 1: wave wv computes h[:, fc + wv*16 .. +16) over K = D ----
        const int fRow = fc + wv * 16 + lid;             // B frag: col n = lid -> w row
        const float* w1p = w1 + ebase + (size_t)fRow * D_DIM + half * 16;
        const float* w3p = w3 + ebase + (size_t)fRow * D_DIM + half * 16;
        // prefetch next chunk's rows (speculative; L2-resident weights)
        __builtin_prefetch(w1p + (size_t)FC * D_DIM, 0, 1);
        __builtin_prefetch(w3p + (size_t)FC * D_DIM, 0, 1);

        v8f a1[2] = {(v8f){}, (v8f){}};
        v8f a3[2] = {(v8f){}, (v8f){}};
        for (int k = 0; k < D_DIM; k += 32) {
            // one weight-fragment pair feeds both row-tiles (2x WMMA per load)
            const v16bf b1 = load16_row_bf16(w1p + k);
            const v16bf b3 = load16_row_bf16(w3p + k);
            #pragma unroll
            for (int r2 = 0; r2 < 2; ++r2) {
                const v16bf av =
                    *(const v16bf*)&Xs[(r2 * 16 + lid) * XS_STRIDE + k + half * 16];
                a1[r2] = __builtin_amdgcn_wmma_f32_16x16x32_bf16(false, av, false, b1,
                                                                 (short)0, a1[r2],
                                                                 false, false);
                a3[r2] = __builtin_amdgcn_wmma_f32_16x16x32_bf16(false, av, false, b3,
                                                                 (short)0, a3[r2],
                                                                 false, false);
            }
        }
        // silu(a1) * a3 -> Hs   (C layout: m = r2*16 + i + half*8, n = lid)
        #pragma unroll
        for (int r2 = 0; r2 < 2; ++r2)
            #pragma unroll
            for (int i = 0; i < 8; ++i) {
                const float v  = a1[r2][i];
                const float s  = __builtin_amdgcn_rcpf(1.f + __expf(-v));
                const float hv = v * s * a3[r2][i];
                Hs[(r2 * 16 + i + half * 8) * HS_STRIDE + wv * 16 + lid] = (__bf16)hv;
            }
        __syncthreads();

        // ---- stage 2: out[:, wv*128 .. +128) += H_chunk @ W2_chunk ----
        for (int kk = 0; kk < FC; kk += 32) {
            // cooperative transposed staging of w2[fc+kk .. +32)[0..D) as bf16:
            // thread grabs a k-pair at one d (coalesced in d), packs, one b32 LDS store
            for (int i = tid; i < 16 * D_DIM; i += 256) {
                const int kp = i >> 10;                // 0..15 -> k = 2*kp, 2*kp+1
                const int d  = i & (D_DIM - 1);
                const size_t g = ebase + (size_t)(fc + kk + 2 * kp) * D_DIM + d;
                v2bf p;
                p[0] = (__bf16)w2[g];
                p[1] = (__bf16)w2[g + D_DIM];
                *(v2bf*)&W2sT[d * W2_STRIDE + 2 * kp] = p;
            }
            __syncthreads();

            const v16bf ah0 = *(const v16bf*)&Hs[lid * HS_STRIDE + kk + half * 16];
            const v16bf ah1 = *(const v16bf*)&Hs[(16 + lid) * HS_STRIDE + kk + half * 16];
            #pragma unroll
            for (int j = 0; j < 8; ++j) {
                const int d0 = wv * 128 + j * 16;
                const v16bf bw =
                    *(const v16bf*)&W2sT[(d0 + lid) * W2_STRIDE + half * 16];
                accO[0][j] = __builtin_amdgcn_wmma_f32_16x16x32_bf16(false, ah0, false, bw,
                                                                     (short)0, accO[0][j],
                                                                     false, false);
                accO[1][j] = __builtin_amdgcn_wmma_f32_16x16x32_bf16(false, ah1, false, bw,
                                                                     (short)0, accO[1][j],
                                                                     false, false);
            }
            __syncthreads();   // W2sT / Hs reuse
        }
    }

    // ---- epilogue: out[tok, d] += coef[tok] * acc ----
    #pragma unroll
    for (int r2 = 0; r2 < 2; ++r2)
        #pragma unroll
        for (int j = 0; j < 8; ++j) {
            const int d0 = wv * 128 + j * 16;
            #pragma unroll
            for (int i = 0; i < 8; ++i) {
                const int   m   = r2 * 16 + i + half * 8;
                const float val = accO[r2][j][i] * coef_s[m];
                atomicAdd(&out[(size_t)tok_s[m] * D_DIM + d0 + lid], val);
            }
        }
}

extern "C" void kernel_launch(void* const* d_in, const int* in_sizes, int n_in,
                              void* d_out, int out_size, void* d_ws, size_t ws_size,
                              hipStream_t stream) {
    const float* x  = (const float*)d_in[0];
    const float* gw = (const float*)d_in[1];
    const float* w1 = (const float*)d_in[2];
    const float* w2 = (const float*)d_in[3];
    const float* w3 = (const float*)d_in[4];
    float* out = (float*)d_out;

    // workspace layout: [cnt: 16 ints][idx: E*T ints][coef: E*T floats] (~512 KB)
    int*   cnt   = (int*)d_ws;
    int*   idxl  = cnt + 16;
    float* coefl = (float*)(idxl + E_EXP * T_TOK);

    // opt in to >64KB dynamic LDS (CDNA5: 320KB per workgroup); host-side, capture-safe
    (void)hipFuncSetAttribute((const void*)moe_ffn_kernel,
                              hipFuncAttributeMaxDynamicSharedMemorySize,
                              (int)SMEM_BYTES);

    (void)hipMemsetAsync(cnt, 0, 16 * sizeof(int), stream);
    (void)hipMemsetAsync(out, 0, (size_t)T_TOK * D_DIM * sizeof(float), stream);

    moe_gate_kernel<<<dim3(T_TOK / 256), dim3(256), 0, stream>>>(x, gw, cnt, idxl, coefl);
    moe_ffn_kernel<<<dim3(T_TOK / BM, E_EXP), dim3(256), SMEM_BYTES, stream>>>(
        x, w1, w2, w3, cnt, idxl, coefl, out);
}
